// ConditionalConv2D_3075196584425
// MI455X (gfx1250) — matrix operand order repaired and checked by reading, not compiled, not run on hardware
//
#include <hip/hip_runtime.h>
#include <stdint.h>

// ---------------------------------------------------------------------------
// Conditional (per-class expert) 3x3 SAME conv as implicit GEMM on CDNA5 WMMA.
//   x:       [B=32, H=64, W=64, CIN=128]  f32
//   classes: [B, 1]                       i32
//   kernel:  [NCLS, 3, 3, CIN, F=256]     f32
//   bias:    [NCLS, F]                    f32
//   out:     [B, H, W, F]                 f32
// GEMM view per sample: M=H*W, N=F, K=9*CIN. bf16 inputs, f32 accumulate.
//
// TIER 2 (ws >= padded-x + packed-w): pre-passes convert x to a zero-padded
//   bf16 image [B][66][66][CIN] and weights to bf16 [cls*9+tap][F][CIN].
//   Hot loop stages tiles with global_load_async_to_lds_b128 (ASYNCcnt) —
//   zero conversion VALU, zero bounds masks, no VGPR round-trip.
// TIER 1 (ws >= packed-w): bf16 weights pre-packed; A converted inline.
// TIER 0: everything inline.
// ---------------------------------------------------------------------------

typedef __bf16 bf16;
typedef bf16  v8bf  __attribute__((ext_vector_type(8)));
typedef bf16  v16bf __attribute__((ext_vector_type(16)));
typedef float v8f   __attribute__((ext_vector_type(8)));

#define H_    64
#define W_    64
#define HP_   66      // padded
#define CIN_  128
#define F_    256
#define KHW_  9       // 3x3 taps
#define BM    64      // pixel tile = one full image row (W_ == BM)
#define BN    128     // output-channel tile
#define BK    32      // bf16 K-slice per WMMA step
#define STEPS 36      // 9 taps * (CIN_/BK)
#define APAD  40      // LDS row padding (80B rows -> conflict-free)
#define BPAD  40

// round-to-nearest f32 -> bf16 (packed pair -> one dword)
__device__ __forceinline__ uint32_t pack_bf16x2(float a, float b) {
  uint32_t ua = __float_as_uint(a);
  uint32_t ub = __float_as_uint(b);
  ua = ua + 0x7FFFu + ((ua >> 16) & 1u);
  ub = ub + 0x7FFFu + ((ub >> 16) & 1u);
  return (ua >> 16) | (ub & 0xFFFF0000u);
}
__device__ __forceinline__ unsigned short bf16_1(float a) {
  uint32_t ua = __float_as_uint(a);
  return (unsigned short)((ua + 0x7FFFu + ((ua >> 16) & 1u)) >> 16);
}

// gfx1250 async global -> LDS copy, tracked by ASYNCcnt (ISA 08 §4)
__device__ __forceinline__ void async_copy_b128(void* lds_dst, const void* gsrc) {
  uint32_t l = (uint32_t)(uintptr_t)lds_dst;   // low 32 bits == LDS offset
  asm volatile("global_load_async_to_lds_b128 %0, %1, off"
               :: "v"(l), "v"(gsrc) : "memory");
}
__device__ __forceinline__ void async_wait0() {
  asm volatile("s_wait_asynccnt 0x0" ::: "memory");
}

// ------------------- pre-pass 1: pack + transpose weights -------------------
// wt[tc][f][ci] (bf16) = kern[tc][ci][f] (f32), tc = cls*9 + kh*3 + kw
__global__ __launch_bounds__(256)
void wpack_kernel(const float* __restrict__ kern, uint32_t* __restrict__ wt32,
                  int totalPairs) {
  int p = blockIdx.x * 256 + threadIdx.x;
  if (p >= totalPairs) return;
  int ci2 = p & 63;                 // pair of input channels
  int f   = (p >> 6) & (F_ - 1);
  int tc  = p >> 14;                // cls*9 + tap
  const float* s = kern + ((size_t)tc * CIN_ + 2 * ci2) * F_ + f;
  wt32[((size_t)tc * F_ + f) * (CIN_ / 2) + ci2] = pack_bf16x2(s[0], s[F_]);
}

// ------------------- pre-pass 2: pad + convert x to bf16 --------------------
// xb[b][hp][wp][ci] (bf16), hp/wp in [0,66); border = 0, interior = x[...]
__global__ __launch_bounds__(256)
void xpad_kernel(const float* __restrict__ x, uint32_t* __restrict__ xb32,
                 int totalPairs) {
  int p = blockIdx.x * 256 + threadIdx.x;
  if (p >= totalPairs) return;
  int ci2 = p & 63;
  int q   = p >> 6;                 // (b*66 + hp)*66 + wp
  int wp  = q % HP_;
  int q2  = q / HP_;
  int hp  = q2 % HP_;
  int b   = q2 / HP_;
  bool ok = (hp >= 1) && (hp <= H_) && (wp >= 1) && (wp <= W_);
  int hc  = ok ? hp - 1 : 0;
  int wc  = ok ? wp - 1 : 0;
  const float* s = x + (((size_t)b * H_ + hc) * W_ + wc) * CIN_ + 2 * ci2;
  float a = s[0], c = s[1];
  a = ok ? a : 0.f;
  c = ok ? c : 0.f;
  xb32[(size_t)q * (CIN_ / 2) + ci2] = pack_bf16x2(a, c);
}

// ------------------------------- main kernel --------------------------------
template <int TIER>
__global__ __launch_bounds__(256)
void cond_conv2d_wmma(const float* __restrict__ x,
                      const int*   __restrict__ classes,
                      const float* __restrict__ kern,
                      const float* __restrict__ bias,
                      const unsigned short* __restrict__ xbf,
                      const unsigned short* __restrict__ wt,
                      float*       __restrict__ out) {
  // ping-pong tiles: A [pixel][k] bf16, B [n][k] bf16
  __shared__ __align__(16) unsigned short ldsA[2][BM][APAD];
  __shared__ __align__(16) unsigned short ldsB[2][BN][BPAD];

  const int h   = blockIdx.x;                  // output row; pixels are w=0..63
  const int n0  = blockIdx.y * BN;             // output-channel tile base
  const int b   = blockIdx.z;                  // sample
  const int cls9 = classes[b] * KHW_;          // per-sample expert base tap

  const int tid   = threadIdx.x;
  const int lane  = tid & 31;                  // wave32
  const int wave  = tid >> 5;                  // 0..7
  const int waveM = wave >> 2;                 // 0..1 : 32-pixel band
  const int waveN = wave & 3;                  // 0..3 : 32-channel band
  const int hl    = lane >> 4;                 // lane half (ISA frag layout)
  const int lr    = lane & 15;

  const v8f vzero = {0.f, 0.f, 0.f, 0.f, 0.f, 0.f, 0.f, 0.f};
  v8f acc[2][2];
#pragma unroll
  for (int mt = 0; mt < 2; ++mt)
#pragma unroll
    for (int nt = 0; nt < 2; ++nt) acc[mt][nt] = vzero;

  // A staging: 4 threads per pixel row, 8 channels (16B) each
  const int ar  = tid >> 2;                    // pixel (== w) 0..63
  const int aci = (tid & 3) * 8;               // ci offset within BK chunk
  // B staging: 1 thread per n-row half (32B per thread)
  const int nB  = tid & (BN - 1);
  const int hb  = tid >> 7;                    // 0/1: which 16-k half

  // ----- stage step s into LDS buffer `buf` --------------------------------
  auto stage = [&](int s, int buf) {
    const int kh = s / 12;
    const int kw = (s >> 2) % 3;
    const int c0 = (s & 3) * BK;
    const int tc = cls9 + kh * 3 + kw;

    if constexpr (TIER == 2) {
      // ---- A: padded bf16 image -> LDS, pure async b128 copy ----
      const unsigned short* asrc =
          xbf + (((size_t)b * HP_ + (h + kh)) * HP_ + (ar + kw)) * CIN_ + c0 + aci;
      async_copy_b128(&ldsA[buf][ar][aci], asrc);
      // ---- B: packed bf16 weights -> LDS, two async b128 copies ----
      const unsigned short* bsrc =
          wt + ((size_t)tc * F_ + n0 + nB) * CIN_ + c0 + hb * 16;
      async_copy_b128(&ldsB[buf][nB][hb * 16],     bsrc);
      async_copy_b128(&ldsB[buf][nB][hb * 16 + 8], bsrc + 8);
    } else {
      // ---- A: clamped-address f32 load, cndmask zeroing, bf16 pack ----
      const int  hin = h + kh - 1;
      const int  win = ar + kw - 1;
      const bool ok  = ((unsigned)hin < (unsigned)H_) && ((unsigned)win < (unsigned)W_);
      const int  hc  = ok ? hin : 0;
      const int  wc  = ok ? win : 0;
      const float4* ap = reinterpret_cast<const float4*>(
          x + (((size_t)b * H_ + hc) * W_ + wc) * CIN_ + c0 + aci);
      float4 f0 = ap[0], f1 = ap[1];
      f0.x = ok ? f0.x : 0.f;  f0.y = ok ? f0.y : 0.f;
      f0.z = ok ? f0.z : 0.f;  f0.w = ok ? f0.w : 0.f;
      f1.x = ok ? f1.x : 0.f;  f1.y = ok ? f1.y : 0.f;
      f1.z = ok ? f1.z : 0.f;  f1.w = ok ? f1.w : 0.f;
      uint4 apk;
      apk.x = pack_bf16x2(f0.x, f0.y);
      apk.y = pack_bf16x2(f0.z, f0.w);
      apk.z = pack_bf16x2(f1.x, f1.y);
      apk.w = pack_bf16x2(f1.z, f1.w);
      *reinterpret_cast<uint4*>(&ldsA[buf][ar][aci]) = apk;

      if constexpr (TIER == 1) {
        // packed bf16 weights: straight coalesced b128 copies
        const uint4* bp = reinterpret_cast<const uint4*>(
            wt + ((size_t)tc * F_ + n0 + nB) * CIN_ + c0 + hb * 16);
        uint4 q0 = bp[0];
        uint4 q1 = bp[1];
        *reinterpret_cast<uint4*>(&ldsB[buf][nB][hb * 16])     = q0;
        *reinterpret_cast<uint4*>(&ldsB[buf][nB][hb * 16 + 8]) = q1;
      } else {
        // fully inline: f32 loads (float4 along F), transpose via b16 stores
        const int n4 = (tid & 31) * 4;
        const int kq = tid >> 5;               // 0..7
        const float* kb = kern + ((size_t)tc * CIN_ + c0) * F_ + n0 + n4;
#pragma unroll
        for (int j = 0; j < 4; ++j) {
          const int k = kq * 4 + j;
          float4 v = *reinterpret_cast<const float4*>(kb + (size_t)k * F_);
          ldsB[buf][n4 + 0][k] = bf16_1(v.x);
          ldsB[buf][n4 + 1][k] = bf16_1(v.y);
          ldsB[buf][n4 + 2][k] = bf16_1(v.z);
          ldsB[buf][n4 + 3][k] = bf16_1(v.w);
        }
      }
    }
  };

  // ------------------------- pipelined main loop ---------------------------
  stage(0, 0);
  for (int s = 0; s < STEPS; ++s) {
    const int buf = s & 1;
    if constexpr (TIER == 2) async_wait0();   // our async stores landed in LDS
    __syncthreads();               // step-s tiles visible; step-(s-1) compute done
    if (s + 1 < STEPS) stage(s + 1, buf ^ 1); // overlaps with WMMAs below

    // fragments per ISA 7.12.2 bf16 layouts
    v16bf afrag[2], bfrag[2];
#pragma unroll
    for (int mt = 0; mt < 2; ++mt) {
      int row = waveM * 32 + mt * 16 + lr;               // M = lane%16
      v8bf lo = *reinterpret_cast<const v8bf*>(&ldsA[buf][row][hl * 8]);
      v8bf hi = *reinterpret_cast<const v8bf*>(&ldsA[buf][row][16 + hl * 8]);
      afrag[mt] = __builtin_shufflevector(lo, hi, 0, 1, 2, 3, 4, 5, 6, 7,
                                                  8, 9, 10, 11, 12, 13, 14, 15);
    }
#pragma unroll
    for (int nt = 0; nt < 2; ++nt) {
      int col = waveN * 32 + nt * 16 + lr;               // N = lane%16
      v8bf lo = *reinterpret_cast<const v8bf*>(&ldsB[buf][col][hl * 8]);
      v8bf hi = *reinterpret_cast<const v8bf*>(&ldsB[buf][col][16 + hl * 8]);
      bfrag[nt] = __builtin_shufflevector(lo, hi, 0, 1, 2, 3, 4, 5, 6, 7,
                                                  8, 9, 10, 11, 12, 13, 14, 15);
    }
#pragma unroll
    for (int mt = 0; mt < 2; ++mt)
#pragma unroll
      for (int nt = 0; nt < 2; ++nt)
        acc[mt][nt] = __builtin_amdgcn_wmma_f32_16x16x32_bf16(
            false, afrag[mt], false, bfrag[nt],
            (short)0, acc[mt][nt], false, false);
  }

  // ---------------- epilogue: + bias[cls], coalesced f32 stores ------------
  const int cls = cls9 / KHW_;
  float bv[2];
#pragma unroll
  for (int nt = 0; nt < 2; ++nt)
    bv[nt] = bias[(size_t)cls * F_ + n0 + waveN * 32 + nt * 16 + lr];

  float* orow = out + (((size_t)b * H_ + h) * W_) * F_ + n0;
#pragma unroll
  for (int mt = 0; mt < 2; ++mt) {
#pragma unroll
    for (int r = 0; r < 8; ++r) {
      // C/D layout: VGPR r -> M = r (lanes 0-15) or r+8 (lanes 16-31)
      int w = waveM * 32 + mt * 16 + hl * 8 + r;
#pragma unroll
      for (int nt = 0; nt < 2; ++nt) {
        orow[(size_t)w * F_ + waveN * 32 + nt * 16 + lr] = acc[mt][nt][r] + bv[nt];
      }
    }
  }
}

extern "C" void kernel_launch(void* const* d_in, const int* in_sizes, int n_in,
                              void* d_out, int out_size, void* d_ws, size_t ws_size,
                              hipStream_t stream) {
  const float* x       = (const float*)d_in[0];
  const int*   classes = (const int*)  d_in[1];
  const float* kern    = (const float*)d_in[2];
  const float* bias    = (const float*)d_in[3];
  float*       out     = (float*)d_out;

  const int B    = in_sizes[1];                          // classes: B*1 elems
  const int ncls = in_sizes[2] / (KHW_ * CIN_ * F_);     // kernel elems / per-class

  dim3 grid(H_, F_ / BN, B);
  dim3 block(256);

  const size_t wtBytes   = (size_t)ncls * KHW_ * F_ * CIN_ * sizeof(unsigned short);
  const size_t xpadBytes = (size_t)B * HP_ * HP_ * CIN_ * sizeof(unsigned short);

  if (ws_size >= xpadBytes + wtBytes) {
    unsigned short* xbf = (unsigned short*)d_ws;
    unsigned short* wt  = (unsigned short*)((char*)d_ws + xpadBytes);
    const int xPairs = B * HP_ * HP_ * (CIN_ / 2);
    const int wPairs = ncls * KHW_ * F_ * (CIN_ / 2);
    xpad_kernel<<<(xPairs + 255) / 256, 256, 0, stream>>>(x, (uint32_t*)xbf, xPairs);
    wpack_kernel<<<(wPairs + 255) / 256, 256, 0, stream>>>(kern, (uint32_t*)wt, wPairs);
    cond_conv2d_wmma<2><<<grid, block, 0, stream>>>(
        x, classes, kern, bias, xbf, wt, out);
  } else if (ws_size >= wtBytes) {
    unsigned short* wt = (unsigned short*)d_ws;
    const int wPairs = ncls * KHW_ * F_ * (CIN_ / 2);
    wpack_kernel<<<(wPairs + 255) / 256, 256, 0, stream>>>(kern, (uint32_t*)wt, wPairs);
    cond_conv2d_wmma<1><<<grid, block, 0, stream>>>(
        x, classes, kern, bias, nullptr, wt, out);
  } else {
    cond_conv2d_wmma<0><<<grid, block, 0, stream>>>(
        x, classes, kern, bias, nullptr, nullptr, out);
  }
}
